// LinearFLH_59579786330516
// MI455X (gfx1250) — matrix operand order repaired and checked by reading, not compiled, not run on hardware
//
#include <hip/hip_runtime.h>

typedef __attribute__((ext_vector_type(16))) _Float16 v16h;
typedef __attribute__((ext_vector_type(8)))  _Float16 v8h;
typedef __attribute__((ext_vector_type(2)))  _Float16 v2h;
typedef __attribute__((ext_vector_type(8)))  float    v8f;

#define OUT_F   11008
#define IN_F    4096
#define NGROUPS 32
#define TOKENS  4096

#define BM 128
#define BN 128
#define BK 64
#define LDS_K (BK + 8)            // 72 halves = 144B stride -> conflict-free frag reads
#define WS_LDS_HALF_OFF (BM * LDS_K)   // Ws starts after Xs in the single shared buffer

#define USE_ASYNC_LDS 1           // global_load_async_to_lds_b128 path (verified to assemble)

// ---------------- kernel 1: one-shot W int4 -> f16 dequant ----------------
// w_packed: int32 elements, each holding one byte = 2 nibbles (lo nibble = even k)
__global__ __launch_bounds__(256)
void w4_dequant_to_f16(const int*   __restrict__ w_packed,
                       const float* __restrict__ w_scale,
                       const int*   __restrict__ w_zero,
                       _Float16*    __restrict__ wh)
{
    const size_t t  = (size_t)blockIdx.x * 256 + threadIdx.x;
    const size_t w0 = t * 4;                       // 4 packed words -> 8 weights
    const int    n  = (int)(w0 >> 11);             // 2048 words per channel
    const int    g  = ((int)w0 & 2047) >> 6;       // 64 words per quant group
    const float  scale = w_scale[(size_t)n * NGROUPS + g];
    const int    zero  = w_zero [(size_t)n * NGROUPS + g];

    const _Float16 sh = (_Float16)scale;
    const _Float16 bh = (_Float16)(float)(1032 + zero);   // 1024 + 8 + zero (exact in f16)
    const v2h s2 = { sh, sh };
    const v2h b2 = { bh, bh };

    const int4 p = *(const int4*)(w_packed + w0);
    const int  b4[4] = { p.x, p.y, p.z, p.w };
    v8h outv;
    #pragma unroll
    for (int j = 0; j < 4; ++j) {
        unsigned tb = (unsigned)b4[j] ^ 0x88u;                          // nib^8 in both nibbles
        unsigned u  = (tb & 0xFu) | ((tb & 0xF0u) << 12) | 0x64006400u; // {1024+lo', 1024+hi'}
        v2h h = __builtin_bit_cast(v2h, u);
        v2h v = (h - b2) * s2;                                          // (signed - zero) * scale
        outv[2 * j]     = v[0];
        outv[2 * j + 1] = v[1];
    }
    *(v8h*)(wh + w0 * 2) = outv;
}

// ---------------- kernel 2: one-shot X fp32 -> f16 ----------------
__global__ __launch_bounds__(256)
void x_cvt_to_f16(const float* __restrict__ x, _Float16* __restrict__ xh)
{
    const size_t i0 = ((size_t)blockIdx.x * 256 + threadIdx.x) * 8;
    const float4 a = *(const float4*)(x + i0);
    const float4 b = *(const float4*)(x + i0 + 4);
    v8h h = { (_Float16)a.x, (_Float16)a.y, (_Float16)a.z, (_Float16)a.w,
              (_Float16)b.x, (_Float16)b.y, (_Float16)b.z, (_Float16)b.w };
    *(v8h*)(xh + i0) = h;
}

// ---------------- async copy: global -> LDS, 16B per lane ----------------
__device__ __forceinline__ void copy_b128_to_lds(unsigned lds_byte_off,
                                                 const _Float16* base,
                                                 unsigned gbyte_off)
{
#if USE_ASYNC_LDS
    asm volatile("global_load_async_to_lds_b128 %0, %1, %2 offset:0"
                 :
                 : "v"(lds_byte_off), "v"(gbyte_off), "s"(base)
                 : "memory");
#endif
}

__device__ __forceinline__ void wait_async_all()
{
#if USE_ASYNC_LDS
    asm volatile("s_wait_asynccnt 0x0" ::: "memory");
#endif
}

// ---------------- kernel 3: f16 WMMA GEMM ----------------
// C[TOKENS, OUT_F] = Xh[TOKENS, IN_F] * Wh[OUT_F, IN_F]^T
__global__ __launch_bounds__(256, 2)
void f16_wmma_gemm(const _Float16* __restrict__ xh,
                   const _Float16* __restrict__ wh,
                   float*          __restrict__ out)
{
    __shared__ _Float16 smem[(BM + BN) * LDS_K];   // Xs then Ws; 36,864 B, group-seg offset 0

    // Escape the LDS object: the async-copy asm ("memory" clobber) must be seen
    // as potentially writing smem, or the ds_load fragment reads get deleted.
    // NOTE: input-only escape — accesses below still go through `smem` directly
    // so address-space inference keeps them as ds_load (round 3 showed that
    // volatile reads instead degrade to serialized flat SYS loads).
    asm volatile("" :: "v"((const void*)smem) : "memory");

    const int tid    = threadIdx.x;
    const int lane   = tid & 31;
    const int wave   = tid >> 5;       // 0..7
    const int wm     = wave >> 2;      // 0..1 : 64 rows each
    const int wn     = wave & 3;       // 0..3 : 32 cols each
    const int laneLo = lane & 15;
    const int laneHi = lane >> 4;

    const int mBlock = blockIdx.y * BM;
    const int nBlock = blockIdx.x * BN;

    // staging geometry: 1024 16B-chunks per operand, 4 per thread
    // idx = tid + i*256 ; row = idx>>3 (0..127) ; chunk = idx&7 (8 halves each)
    v8f acc[4][2] = {};

    for (int k0 = 0; k0 < IN_F; k0 += BK) {
        #pragma unroll
        for (int i = 0; i < 4; ++i) {
            const int idx   = tid + i * 256;
            const int row   = idx >> 3;
            const int hoff  = (idx & 7) << 3;              // half offset within row
            const unsigned ldsX = (unsigned)((row * LDS_K + hoff) * 2);
            const unsigned ldsW = (unsigned)((WS_LDS_HALF_OFF + row * LDS_K + hoff) * 2);
#if USE_ASYNC_LDS
            copy_b128_to_lds(ldsX, xh,
                (unsigned)(((size_t)(mBlock + row) * IN_F + k0 + hoff) * 2));
            copy_b128_to_lds(ldsW, wh,
                (unsigned)(((size_t)(nBlock + row) * IN_F + k0 + hoff) * 2));
#else
            *(v8h*)((char*)smem + ldsX) =
                *(const v8h*)(xh + (size_t)(mBlock + row) * IN_F + k0 + hoff);
            *(v8h*)((char*)smem + ldsW) =
                *(const v8h*)(wh + (size_t)(nBlock + row) * IN_F + k0 + hoff);
#endif
        }
        wait_async_all();
        __syncthreads();

        // ---- compute: 2 WMMA K-steps of 32 ----
        #pragma unroll
        for (int kk = 0; kk < 2; ++kk) {
            const int kOff = kk * 32;

            // A frags (16x32 f16): lane = row M; lo lanes K{0..7,16..23}, hi lanes K{8..15,24..31}
            v16h afr[4];
            #pragma unroll
            for (int tm = 0; tm < 4; ++tm) {
                const _Float16* ap =
                    &smem[(wm * 64 + tm * 16 + laneLo) * LDS_K + kOff + laneHi * 8];
                const v8h a0 = *(const v8h*)(ap);
                const v8h a1 = *(const v8h*)(ap + 16);
                afr[tm] = __builtin_shufflevector(a0, a1,
                          0,1,2,3,4,5,6,7,8,9,10,11,12,13,14,15);
            }
            // B frags (32x16 f16): lane = col N; lo lanes K{0..7,16..23}, hi lanes K{8..15,24..31}
            #pragma unroll
            for (int tn = 0; tn < 2; ++tn) {
                const _Float16* bp =
                    &smem[WS_LDS_HALF_OFF +
                          (wn * 32 + tn * 16 + laneLo) * LDS_K + kOff + laneHi * 8];
                const v8h b0 = *(const v8h*)(bp);
                const v8h b1 = *(const v8h*)(bp + 16);
                const v16h bfr = __builtin_shufflevector(b0, b1,
                          0,1,2,3,4,5,6,7,8,9,10,11,12,13,14,15);
                #pragma unroll
                for (int tm = 0; tm < 4; ++tm) {
                    acc[tm][tn] = __builtin_amdgcn_wmma_f32_16x16x32_f16(
                        false, afr[tm], false, bfr,
                        (short)0, acc[tm][tn], false, false);
                }
            }
        }
        __syncthreads();
    }

    // ---- epilogue: VGPR r -> M = r (+8 hi lanes), N = laneLo ----
    #pragma unroll
    for (int tm = 0; tm < 4; ++tm) {
        const int rowBase = mBlock + wm * 64 + tm * 16 + laneHi * 8;
        #pragma unroll
        for (int tn = 0; tn < 2; ++tn) {
            const int col = nBlock + wn * 32 + tn * 16 + laneLo;
            #pragma unroll
            for (int r = 0; r < 8; ++r) {
                out[(size_t)(rowBase + r) * OUT_F + col] = acc[tm][tn][r];
            }
        }
    }
}

extern "C" void kernel_launch(void* const* d_in, const int* in_sizes, int n_in,
                              void* d_out, int out_size, void* d_ws, size_t ws_size,
                              hipStream_t stream) {
    (void)in_sizes; (void)n_in; (void)out_size; (void)ws_size;
    const float* xp  = (const float*)d_in[0];
    const int*   wpk = (const int*)d_in[1];
    const float* wsc = (const float*)d_in[2];
    const int*   wzp = (const int*)d_in[3];
    float*       op  = (float*)d_out;

    // workspace: Wh f16 [OUT_F*IN_F] then Xh f16 [TOKENS*IN_F]  (~118 MB)
    _Float16* wh = (_Float16*)d_ws;
    _Float16* xh = (_Float16*)((char*)d_ws + (size_t)OUT_F * IN_F * 2);

    // 1) dequant W (22,544,384 packed words / 4 per thread / 256)
    w4_dequant_to_f16<<<dim3(22016), dim3(256), 0, stream>>>(wpk, wsc, wzp, wh);
    // 2) convert X (16,777,216 f32 / 8 per thread / 256)
    x_cvt_to_f16<<<dim3(8192), dim3(256), 0, stream>>>(xp, xh);
    // 3) f16 WMMA GEMM
    f16_wmma_gemm<<<dim3(OUT_F / BN, TOKENS / BM), dim3(256), 0, stream>>>(xh, wh, op);
}
